// SharedSelfAttentionCLSModel_37477884625074
// MI455X (gfx1250) — matrix operand order repaired and checked by reading, not compiled, not run on hardware
//
#include <hip/hip_runtime.h>
#include <hip/hip_bf16.h>

// ---------------------------------------------------------------------------
// Model dims
// ---------------------------------------------------------------------------
#define Bc   64
#define NTc  456
#define TOKc 64
#define Lc   457      // NT + 1 (cls token)
#define Dc   256
#define Hc   8
#define HDc  32
#define FFc  1024
#define NLc  4
#define MR   (Bc * Lc)       // 29248 rows of x (divisible by 64)
#define MP   (Bc * NTc)      // 29184 rows for input projection (divisible by 64)

typedef __attribute__((ext_vector_type(16))) __bf16 bf16x16;
typedef __attribute__((ext_vector_type(8)))  float  f32x8;

union Frag {
    bf16x16        v;
    uint4          u4[2];
    unsigned short s[16];
};

__device__ __forceinline__ unsigned short f2bf(float f) {
    unsigned u = __builtin_bit_cast(unsigned, f);
    u += 0x7FFFu + ((u >> 16) & 1u);          // round-to-nearest-even
    return (unsigned short)(u >> 16);
}

// ---------------------------------------------------------------------------
// f32 -> bf16 conversion (grid-strided)
// ---------------------------------------------------------------------------
__global__ void k_cvt(const float* __restrict__ s, unsigned short* __restrict__ d, long n) {
    long i = (long)blockIdx.x * blockDim.x + threadIdx.x;
    long stride = (long)gridDim.x * blockDim.x;
    for (long j = i; j < n; j += stride) d[j] = f2bf(s[j]);
}

// ---------------------------------------------------------------------------
// WMMA GEMM:  out(M,N) = A(M,K)bf16 @ W(K,N)bf16 + bias, opt ReLU.
// Block = 128 threads (4 waves).  Block tile = 64 rows x 64 cols:
//   - one LDS-staged transposed W tile (32K x 64N, 4KB) per K-step
//   - each wave loads its A fragment ONCE per K-step and reuses it for
//     4 column sub-tiles -> 4 v_wmma per wave per K-step (16 per block).
// Row remap: outRow = (r / rpbI) * rpbO + rowOff + (r % rpbI)
// (identity fast-path when rpbI >= M).
// Columns beyond N are zero-padded in LDS and skipped at the store.
// ---------------------------------------------------------------------------
__global__ __launch_bounds__(128)
void k_gemm(const unsigned short* __restrict__ A, const unsigned short* __restrict__ W,
            const float* __restrict__ bias, float* __restrict__ outF,
            unsigned short* __restrict__ outB,
            int M, int N, int K, int rpbI, int rpbO, int rowOff, int relu)
{
    __shared__ __align__(16) unsigned short tileT[64][32];   // [n][k], 4KB

    const int t    = threadIdx.x;
    const int w    = t >> 5;
    const int lane = t & 31;
    const int half = lane >> 4;
    const int n    = lane & 15;
    const int nBase = blockIdx.x * 64;
    const int mBase = blockIdx.y * 64 + w * 16;
    const int row   = mBase + n;                 // A-fragment row (lane m)

    f32x8 acc[4];
#pragma unroll
    for (int j = 0; j < 4; j++) acc[j] = f32x8{0.f,0.f,0.f,0.f,0.f,0.f,0.f,0.f};

    const unsigned short* aRow = A + (long)row * K;

    // cooperative-load geometry: thread t covers W[k0+kk][nBase+nn .. +15]
    const int kk = t >> 2;            // 0..31
    const int nn = (t & 3) * 16;      // 0,16,32,48
    const bool colOk = (nBase + nn) < N;   // whole 16-chunk valid or not

    for (int k0 = 0; k0 < K; k0 += 32) {
        __syncthreads();
        {   // stage W[k0..k0+31][nBase..nBase+63] -> tileT[n][k] (transposed)
            uint4 w0 = {0,0,0,0}, w1 = {0,0,0,0};
            if (colOk) {
                const unsigned short* wp = W + (long)(k0 + kk) * N + nBase + nn;
                w0 = *(const uint4*)(wp);
                w1 = *(const uint4*)(wp + 8);
            }
            const unsigned short* ws0 = (const unsigned short*)&w0;
            const unsigned short* ws1 = (const unsigned short*)&w1;
#pragma unroll
            for (int j = 0; j < 8; j++) {
                tileT[nn + j][kk]     = ws0[j];
                tileT[nn + 8 + j][kk] = ws1[j];
            }
        }
        __syncthreads();

        Frag a;
        const unsigned short* ap = aRow + k0;
        __builtin_prefetch(ap + 32, 0, 3);       // near-scope prefetch of next K-step
        a.u4[0] = *(const uint4*)(ap + 8 * half);        // K = 8h .. 8h+7
        a.u4[1] = *(const uint4*)(ap + 16 + 8 * half);   // K = 16+8h .. 16+8h+7

#pragma unroll
        for (int j = 0; j < 4; j++) {
            Frag b;
            b.u4[0] = *(const uint4*)(&tileT[16 * j + n][8 * half]);
            b.u4[1] = *(const uint4*)(&tileT[16 * j + n][16 + 8 * half]);
            acc[j] = __builtin_amdgcn_wmma_f32_16x16x32_bf16(false, a.v, false, b.v,
                                                             (short)0, acc[j], false, false);
        }
    }

    const bool identity = (rpbI >= M);
#pragma unroll
    for (int j = 0; j < 4; j++) {
        const int col = nBase + 16 * j + n;
        if (col >= N) continue;
        const float bv = bias ? bias[col] : 0.f;
#pragma unroll
        for (int r = 0; r < 8; r++) {
            int  mr   = mBase + r + 8 * half;                  // C row = r + 8*half
            long orow = identity ? (long)mr
                                 : (long)(mr / rpbI) * rpbO + rowOff + (mr % rpbI);
            float v = acc[j][r] + bv;
            if (relu) v = fmaxf(v, 0.f);
            long idx = orow * N + col;
            if (outF) outF[idx] = v;
            if (outB) outB[idx] = f2bf(v);
        }
    }
}

// ---------------------------------------------------------------------------
// CLS row: x[b][0][:] = coords[b] @ w_cls + b_cls  (tiny, f32 VALU)
// ---------------------------------------------------------------------------
__global__ void k_cls(const float* __restrict__ coords, const float* __restrict__ wc,
                      const float* __restrict__ bc, float* __restrict__ x,
                      unsigned short* __restrict__ xb)
{
    int b = blockIdx.x, d = threadIdx.x;
    float v = bc[d] + coords[b*3+0]*wc[0*Dc + d]
                    + coords[b*3+1]*wc[1*Dc + d]
                    + coords[b*3+2]*wc[2*Dc + d];
    long idx = (long)b * Lc * Dc + d;
    x[idx]  = v;
    xb[idx] = f2bf(v);
}

// ---------------------------------------------------------------------------
// Flash attention, one wave per (query-16-tile, head, batch).
// qkv rows (bf16, stride 768): [q(0..255) | k(256..511) | v(512..767)]
// Computes S^T = K Q^T via WMMA (so softmax rows live in registers),
// then O^T += V^T P^T via WMMA; writes f32 attn output (B,L,D).
// ---------------------------------------------------------------------------
__global__ __launch_bounds__(32)
void k_attn(const unsigned short* __restrict__ qkv, float* __restrict__ outDelta)
{
    const int lane = threadIdx.x & 31;
    const int half = lane >> 4;
    const int n    = lane & 15;          // query column (both halves same query)
    const int qt   = blockIdx.x;         // 0..28
    const int h    = blockIdx.y;
    const int b    = blockIdx.z;
    const int q0   = qt * 16;
    const int qpos = q0 + n;

    const float slope = exp2f(-floorf((float)h * 0.25f));   // alibi: 1/2^(h//4)
    const float scale = 0.17677669529663687f;               // 1/sqrt(32)

    // Q B-fragment (loop invariant): lane n holds Q[qpos][dim = i + 16*half]
    Frag bq;
    {
        int qr = min(qpos, Lc - 1);
        const unsigned short* qp = qkv + ((long)(b * Lc + qr)) * 768 + h * HDc + 16 * half;
        bq.u4[0] = *(const uint4*)(qp);
        bq.u4[1] = *(const uint4*)(qp + 8);
    }

    f32x8 acc0 = {0.f,0.f,0.f,0.f,0.f,0.f,0.f,0.f};   // O^T dims 0..15
    f32x8 acc1 = {0.f,0.f,0.f,0.f,0.f,0.f,0.f,0.f};   // O^T dims 16..31
    const f32x8 zero = {0.f,0.f,0.f,0.f,0.f,0.f,0.f,0.f};
    float m_run = -1e30f, l_run = 0.f;

    for (int kb = 0; kb < 15; kb++) {                 // 15 * 32 = 480 >= 457 keys
        const int kbase = kb * 32;

        // K A-fragments: lane m=n is a key row, dims contiguous
        Frag ak0, ak1;
        {
            const unsigned short* kp0 = qkv + ((long)(b*Lc + min(kbase +      n, Lc-1)))*768 + 256 + h*HDc;
            const unsigned short* kp1 = qkv + ((long)(b*Lc + min(kbase + 16 + n, Lc-1)))*768 + 256 + h*HDc;
            ak0.u4[0] = *(const uint4*)(kp0 + 16*half);
            ak0.u4[1] = *(const uint4*)(kp0 + 16*half + 8);
            ak1.u4[0] = *(const uint4*)(kp1 + 16*half);
            ak1.u4[1] = *(const uint4*)(kp1 + 16*half + 8);
        }
        f32x8 st0 = __builtin_amdgcn_wmma_f32_16x16x32_bf16(false, ak0.v, false, bq.v, (short)0, zero, false, false);
        f32x8 st1 = __builtin_amdgcn_wmma_f32_16x16x32_bf16(false, ak1.v, false, bq.v, (short)0, zero, false, false);

        // scale + alibi + OOB mask; local max over this lane's 16 keys
        float p0[8], p1[8];
        float mloc = -1e30f;
#pragma unroll
        for (int r = 0; r < 8; r++) {
            int k0i = kbase + r + 8 * half;
            int k1i = k0i + 16;
            float s0 = (k0i < Lc) ? st0[r]*scale - slope*fabsf((float)(qpos - k0i)) : -1e30f;
            float s1 = (k1i < Lc) ? st1[r]*scale - slope*fabsf((float)(qpos - k1i)) : -1e30f;
            p0[r] = s0; p1[r] = s1;
            mloc = fmaxf(mloc, fmaxf(s0, s1));
        }
        mloc = fmaxf(mloc, __shfl_xor(mloc, 16, 32));     // merge key halves
        float m_new = fmaxf(m_run, mloc);
        float corr  = expf(m_run - m_new);

        float lsum = 0.f;
#pragma unroll
        for (int r = 0; r < 8; r++) {
            p0[r] = expf(p0[r] - m_new);
            p1[r] = expf(p1[r] - m_new);
            lsum += p0[r] + p1[r];
        }
        lsum += __shfl_xor(lsum, 16, 32);
        l_run = l_run * corr + lsum;
        m_run = m_new;
#pragma unroll
        for (int j = 0; j < 8; j++) { acc0[j] = acc0[j] * corr; acc1[j] = acc1[j] * corr; }

        // Build P^T B-fragment: lane n holds P[query n][key = i + 16*half].
        // Keys are split across lane halves in C-layout -> one xor-16 shuffle.
        Frag pb;
#pragma unroll
        for (int i = 0; i < 8; i++) {
            float s0 = __shfl_xor(p0[i], 16, 32);
            float s1 = __shfl_xor(p1[i], 16, 32);
            float lo = half ? s1   : p0[i];   // element i   -> key i   + 16*half
            float hi = half ? p1[i] : s0;     // element i+8 -> key i+8 + 16*half
            pb.s[i]     = f2bf(lo);
            pb.s[i + 8] = f2bf(hi);
        }

        // V^T A-fragments: lane m=n is a dim, keys strided (per-lane u16 loads)
        Frag av0, av1;
#pragma unroll
        for (int i = 0; i < 16; i++) {
            int kk = kbase + i + 16 * half;
            const unsigned short* vp = qkv + ((long)(b*Lc + min(kk, Lc-1)))*768 + 512 + h*HDc;
            av0.s[i] = vp[n];
            av1.s[i] = vp[16 + n];
        }
        acc0 = __builtin_amdgcn_wmma_f32_16x16x32_bf16(false, av0.v, false, pb.v, (short)0, acc0, false, false);
        acc1 = __builtin_amdgcn_wmma_f32_16x16x32_bf16(false, av1.v, false, pb.v, (short)0, acc1, false, false);
    }

    if (qpos < Lc) {
        float inv = 1.0f / l_run;
        float* op = outDelta + ((long)(b * Lc + qpos)) * Dc + h * HDc;
#pragma unroll
        for (int r = 0; r < 8; r++) {
            op[r + 8 * half]      = acc0[r] * inv;
            op[16 + r + 8 * half] = acc1[r] * inv;
        }
    }
}

// ---------------------------------------------------------------------------
// x = LayerNorm(x + delta); also refresh bf16 mirror. One block per row.
// ---------------------------------------------------------------------------
__global__ __launch_bounds__(256)
void k_add_ln(float* __restrict__ x, const float* __restrict__ d,
              const float* __restrict__ g, const float* __restrict__ bb,
              unsigned short* __restrict__ xb)
{
    __shared__ float red[256];
    const int  t   = threadIdx.x;
    const long row = blockIdx.x;
    float v = x[row * Dc + t] + d[row * Dc + t];

    red[t] = v; __syncthreads();
    for (int s = 128; s > 0; s >>= 1) { if (t < s) red[t] += red[t + s]; __syncthreads(); }
    float mean = red[0] * (1.0f / Dc);
    __syncthreads();

    float c = v - mean;
    red[t] = c * c; __syncthreads();
    for (int s = 128; s > 0; s >>= 1) { if (t < s) red[t] += red[t + s]; __syncthreads(); }
    float var = red[0] * (1.0f / Dc);

    float y = c * rsqrtf(var + 1e-5f) * g[t] + bb[t];
    x[row * Dc + t]  = y;
    xb[row * Dc + t] = f2bf(y);
}

// ---------------------------------------------------------------------------
// Head: BN -> GELU -> dot th2 -> softmax over L -> weighted pool of x.
// One block per batch element.
// ---------------------------------------------------------------------------
__global__ __launch_bounds__(256)
void k_head(const float* __restrict__ hh, const float* __restrict__ x,
            const float* __restrict__ bn_g, const float* __restrict__ bn_b,
            const float* __restrict__ w2, const float* __restrict__ b2,
            float* __restrict__ out)
{
    __shared__ float sc[Lc];
    __shared__ float red[256];
    const int t = threadIdx.x, b = blockIdx.x;
    const float bnScale = rsqrtf(1.0f + 1e-5f);

    float lmax = -1e30f;
    for (int l = t; l < Lc; l += 256) {
        const float* hp = hh + ((long)(b * Lc + l)) * 32;
        float s = b2[0];
#pragma unroll
        for (int j = 0; j < 32; j++) {
            float hv = hp[j] * (bn_g[j] * bnScale) + bn_b[j];
            float u  = 0.7978845608028654f * (hv + 0.044715f * hv * hv * hv);
            s += 0.5f * hv * (1.0f + tanhf(u)) * w2[j];      // tanh-GELU
        }
        sc[l] = s;
        lmax  = fmaxf(lmax, s);
    }
    red[t] = lmax; __syncthreads();
    for (int s = 128; s > 0; s >>= 1) { if (t < s) red[t] = fmaxf(red[t], red[t + s]); __syncthreads(); }
    float mx = red[0]; __syncthreads();

    float lsum = 0.f;
    for (int l = t; l < Lc; l += 256) { float e = expf(sc[l] - mx); sc[l] = e; lsum += e; }
    red[t] = lsum; __syncthreads();
    for (int s = 128; s > 0; s >>= 1) { if (t < s) red[t] += red[t + s]; __syncthreads(); }
    float inv = 1.0f / red[0];
    __syncthreads();

    float accp = 0.f;
    const float* xr = x + (long)b * Lc * Dc;
    for (int l = 0; l < Lc; l++) accp += sc[l] * inv * xr[(long)l * Dc + t];
    out[b * Dc + t] = accp;
}

// ---------------------------------------------------------------------------
// Host orchestration
// ---------------------------------------------------------------------------
extern "C" void kernel_launch(void* const* d_in, const int* in_sizes, int n_in,
                              void* d_out, int out_size, void* d_ws, size_t ws_size,
                              hipStream_t stream)
{
    (void)in_sizes; (void)n_in; (void)out_size; (void)ws_size;
    const float* gene   = (const float*)d_in[0];
    const float* coords = (const float*)d_in[1];
    const float* w_in   = (const float*)d_in[2];
    const float* b_in   = (const float*)d_in[3];
    const float* w_cls  = (const float*)d_in[4];
    const float* b_cls  = (const float*)d_in[5];
    const float* qkv_w  = (const float*)d_in[6];
    const float* qkv_b  = (const float*)d_in[7];
    const float* ln1_g  = (const float*)d_in[8];
    const float* ln1_b  = (const float*)d_in[9];
    const float* ffn_w1 = (const float*)d_in[10];
    const float* ffn_b1 = (const float*)d_in[11];
    const float* ffn_w2 = (const float*)d_in[12];
    const float* ffn_b2 = (const float*)d_in[13];
    const float* ln2_g  = (const float*)d_in[14];
    const float* ln2_b  = (const float*)d_in[15];
    const float* th1_w  = (const float*)d_in[16];
    const float* th1_b  = (const float*)d_in[17];
    const float* bn_g   = (const float*)d_in[18];
    const float* bn_b   = (const float*)d_in[19];
    const float* th2_w  = (const float*)d_in[20];
    const float* th2_b  = (const float*)d_in[21];
    float* out = (float*)d_out;

    // workspace carve (256B aligned)
    char*  ws  = (char*)d_ws;
    size_t off = 0;
    auto carve = [&](size_t bytes) -> char* {
        char* p = ws + off; off += (bytes + 255) & ~size_t(255); return p;
    };
    float*          xF    = (float*)carve((size_t)MR * Dc * 4);
    unsigned short* xB    = (unsigned short*)carve((size_t)MR * Dc * 2);
    unsigned short* qkvB  = (unsigned short*)carve((size_t)MR * 768 * 2);
    unsigned short* hB    = (unsigned short*)carve((size_t)MR * FFc * 2);
    float*          delta = (float*)carve((size_t)MR * Dc * 4);
    float*          hhF   = (float*)carve((size_t)MR * 32 * 4);
    unsigned short* geneB = (unsigned short*)carve((size_t)Bc * NTc * TOKc * 2);
    unsigned short* winB  = (unsigned short*)carve((size_t)TOKc * Dc * 2);
    unsigned short* qwB   = (unsigned short*)carve((size_t)NLc * Dc * 3 * Dc * 2);
    unsigned short* f1B   = (unsigned short*)carve((size_t)NLc * Dc * FFc * 2);
    unsigned short* f2B   = (unsigned short*)carve((size_t)NLc * FFc * Dc * 2);
    unsigned short* t1B   = (unsigned short*)carve((size_t)Dc * 32 * 2);

    auto cvt = [&](const float* s, unsigned short* d, long n) {
        long blk = (n + 255) / 256; if (blk > 4096) blk = 4096;
        k_cvt<<<(unsigned)blk, 256, 0, stream>>>(s, d, n);
    };
    cvt(gene,   geneB, (long)Bc * NTc * TOKc);
    cvt(w_in,   winB,  (long)TOKc * Dc);
    cvt(qkv_w,  qwB,   (long)NLc * Dc * 3 * Dc);
    cvt(ffn_w1, f1B,   (long)NLc * Dc * FFc);
    cvt(ffn_w2, f2B,   (long)NLc * FFc * Dc);
    cvt(th1_w,  t1B,   (long)Dc * 32);

    // input projection: rows (b*456+t) -> x rows (b*457 + 1 + t)
    k_gemm<<<dim3((Dc+63)/64, MP/64), 128, 0, stream>>>(geneB, winB, b_in, xF, xB,
                                                        MP, Dc, TOKc, NTc, Lc, 1, 0);
    k_cls<<<Bc, Dc, 0, stream>>>(coords, w_cls, b_cls, xF, xB);

    for (int l = 0; l < NLc; l++) {
        k_gemm<<<dim3((768+63)/64, MR/64), 128, 0, stream>>>(xB, qwB + (long)l*Dc*3*Dc,
                                                             qkv_b + l*3*Dc, nullptr, qkvB,
                                                             MR, 3*Dc, Dc, MR, MR, 0, 0);
        k_attn<<<dim3(29, Hc, Bc), 32, 0, stream>>>(qkvB, delta);
        k_add_ln<<<MR, 256, 0, stream>>>(xF, delta, ln1_g + l*Dc, ln1_b + l*Dc, xB);

        k_gemm<<<dim3((FFc+63)/64, MR/64), 128, 0, stream>>>(xB, f1B + (long)l*Dc*FFc,
                                                             ffn_b1 + l*FFc, nullptr, hB,
                                                             MR, FFc, Dc, MR, MR, 0, 1);
        k_gemm<<<dim3((Dc+63)/64, MR/64), 128, 0, stream>>>(hB, f2B + (long)l*FFc*Dc,
                                                            ffn_b2 + l*Dc, delta, nullptr,
                                                            MR, Dc, FFc, MR, MR, 0, 0);
        k_add_ln<<<MR, 256, 0, stream>>>(xF, delta, ln2_g + l*Dc, ln2_b + l*Dc, xB);
    }

    k_gemm<<<dim3(1, MR/64), 128, 0, stream>>>(xB, t1B, th1_b, hhF, nullptr,
                                               MR, 32, Dc, MR, MR, 0, 0);
    k_head<<<Bc, 256, 0, stream>>>(hhF, xF, bn_g, bn_b, th2_w, th2_b, out);
}
